// Attention_6021544149033
// MI455X (gfx1250) — matrix operand order repaired and checked by reading, not compiled, not run on hardware
//
#include <hip/hip_runtime.h>
#include <hip/hip_bf16.h>
#include <cstdint>

// ---------------- problem constants (fixed shapes) ----------------
constexpr int BATCH   = 2;
constexpr int NSEQ    = 2048;
constexpr int DIM     = 1024;
constexpr int HEADS   = 8;
constexpr int DH      = 64;
constexpr int INNER   = HEADS * DH;     // 512
constexpr float SCALE = 0.125f;         // 64^-0.5
constexpr int ROWS    = BATCH * NSEQ;   // 4096
constexpr float NEG   = -3.0e38f;

// ---------------- vector types for WMMA ----------------
typedef __bf16 v16bf __attribute__((ext_vector_type(16)));
typedef __bf16 v8bf  __attribute__((ext_vector_type(8)));
typedef float  v8f   __attribute__((ext_vector_type(8)));

__device__ __forceinline__ v8f wmma_bf16(v16bf a, v16bf b, v8f c) {
  // D = A(16x32 bf16) * B(32x16 bf16) + C(16x16 f32)
  return __builtin_amdgcn_wmma_f32_16x16x32_bf16(
      /*neg_a=*/false, a, /*neg_b=*/false, b,
      /*c_mod=*/(short)0, c, /*reuse_a=*/false, /*reuse_b=*/false);
}

// A-fragment (16x32, 16-bit): lane holds row m=lane&15; with b = 16*(lane>>4),
// elements 0..15 -> K=b..b+15 (contiguous).
// `rowptr` points at A[row*lda + k0]; works for global or LDS (generic) ptrs.
__device__ __forceinline__ v16bf load_a_frag(const __bf16* rowptr, int half) {
  const int base = half * 16;
  v8bf lo = *reinterpret_cast<const v8bf*>(rowptr + base);
  v8bf hi = *reinterpret_cast<const v8bf*>(rowptr + base + 8);
  v16bf r;
#pragma unroll
  for (int e = 0; e < 8; ++e) { r[e] = lo[e]; r[e + 8] = hi[e]; }
  return r;
}

__device__ __forceinline__ v8f zero8() {
  v8f z;
#pragma unroll
  for (int e = 0; e < 8; ++e) z[e] = 0.0f;
  return z;
}

// ---------------- kernel 1: LayerNorm + bf16 casts ----------------
__global__ __launch_bounds__(256)
void ln_cast_kernel(const float* __restrict__ x, const float* __restrict__ gamma,
                    __bf16* __restrict__ xnbf, __bf16* __restrict__ xbf) {
  const int row = blockIdx.x;        // 0..ROWS-1
  const int tid = threadIdx.x;       // 0..255
  const float* xr = x + (size_t)row * DIM;
  float s = 0.f, s2 = 0.f;
#pragma unroll
  for (int i = 0; i < DIM / 256; ++i) {
    float v = xr[tid + 256 * i];
    s += v; s2 += v * v;
  }
  __shared__ float sh1[256], sh2[256];
  sh1[tid] = s; sh2[tid] = s2;
  __syncthreads();
  for (int off = 128; off > 0; off >>= 1) {
    if (tid < off) { sh1[tid] += sh1[tid + off]; sh2[tid] += sh2[tid + off]; }
    __syncthreads();
  }
  const float mu   = sh1[0] * (1.0f / DIM);
  const float var  = sh2[0] * (1.0f / DIM) - mu * mu;
  const float rstd = rsqrtf(var + 1e-5f);
#pragma unroll
  for (int i = 0; i < DIM / 256; ++i) {
    const int c = tid + 256 * i;
    const float v = xr[c];
    xnbf[(size_t)row * DIM + c] = (__bf16)((v - mu) * rstd * gamma[c]);
    xbf [(size_t)row * DIM + c] = (__bf16)v;
  }
}

// ---------------- kernel 2: transpose f32 -> bf16 (weights) ----------------
__global__ __launch_bounds__(256)
void transpose_cast_kernel(const float* __restrict__ src, __bf16* __restrict__ dst,
                           int rows, int cols) {
  const int total = rows * cols;
  for (int idx = blockIdx.x * blockDim.x + threadIdx.x; idx < total;
       idx += gridDim.x * blockDim.x) {
    const int r = idx / cols, c = idx % cols;
    dst[(size_t)c * rows + r] = (__bf16)src[idx];
  }
}

// ---------------- kernel 3: generic bf16 WMMA GEMM, wave = 16x64 tile -------
// A: [M][K] bf16 row-major.  Bt: [Ncols][K] bf16 (pre-transposed B).
// MODE 0: out0 = bf16 [M][Ncols], scaled by SCALE          (Q projection)
// MODE 1: Ncols=128: cols 0..63 -> out0 = k [M][64] bf16;
//         cols 64..127 -> out1 = vT [b][64][NSEQ] bf16     (KV projection)
// MODE 2: out0 = f32 [M][Ncols]                            (output projection)
template <int MODE>
__global__ __launch_bounds__(32)
void gemm_wmma_kernel(const __bf16* __restrict__ A, const __bf16* __restrict__ Bt,
                      int K, int Ncols, void* __restrict__ out0,
                      void* __restrict__ out1) {
  const int lane = threadIdx.x;
  const int half = lane >> 4;
  const int ln   = lane & 15;
  const int m0   = blockIdx.x * 16;
  const int n0   = blockIdx.y * 64;

  v8f acc[4];
#pragma unroll
  for (int j = 0; j < 4; ++j) acc[j] = zero8();

  const __bf16* arow = A + (size_t)(m0 + ln) * K;
  for (int k0 = 0; k0 < K; k0 += 32) {
    const v16bf af = load_a_frag(arow + k0, half);
#pragma unroll
    for (int j = 0; j < 4; ++j) {
      const __bf16* bp = Bt + (size_t)(n0 + j * 16 + ln) * K + k0 + half * 16;
      const v16bf bf_ = *reinterpret_cast<const v16bf*>(bp);
      acc[j] = wmma_bf16(af, bf_, acc[j]);
    }
  }

#pragma unroll
  for (int j = 0; j < 4; ++j) {
#pragma unroll
    for (int r = 0; r < 8; ++r) {
      const int m = m0 + r + half * 8;          // C layout: VGPR r, half -> M
      const int c = n0 + j * 16 + ln;           // N = lane&15
      const float v = acc[j][r];
      if (MODE == 0) {
        ((__bf16*)out0)[(size_t)m * Ncols + c] = (__bf16)(v * SCALE);
      } else if (MODE == 1) {
        if (c < DH) {
          ((__bf16*)out0)[(size_t)m * DH + c] = (__bf16)v;           // K
        } else {
          const int b  = m >> 11;               // m / NSEQ
          const int nr = m & (NSEQ - 1);
          ((__bf16*)out1)[((size_t)b * DH + (c - DH)) * NSEQ + nr] = (__bf16)v; // V^T
        }
      } else {
        ((float*)out0)[(size_t)m * Ncols + c] = v;
      }
    }
  }
}

// ---------------- kernel 4: flash attention, wave = (b,h, 16 q rows) --------
__global__ __launch_bounds__(32)
void flash_kernel(const __bf16* __restrict__ qbf,   // [B*N][INNER], pre-scaled
                  const __bf16* __restrict__ kbf,   // [B*N][DH]
                  const __bf16* __restrict__ vt,    // [B][DH][N]
                  const float*  __restrict__ bias,  // [H][N][N]
                  const unsigned char* __restrict__ mask, // [B][N] bool
                  __bf16* __restrict__ obf) {       // [B*N][INNER]
  const int lane = threadIdx.x;
  const int half = lane >> 4;
  const int ln   = lane & 15;
  const int i0   = blockIdx.x * 16;
  const int h    = blockIdx.y;
  const int b    = blockIdx.z;

  __shared__ __bf16 pls[16 * 32];   // P tile staging (C-layout -> A-layout)

  // Q A-fragments for K-dim chunks d=0..31 and d=32..63
  const __bf16* qrow = qbf + (size_t)(b * NSEQ + i0 + ln) * INNER + h * DH;
  const v16bf aq0 = load_a_frag(qrow, half);
  const v16bf aq1 = load_a_frag(qrow + 32, half);

  v8f o[4];
#pragma unroll
  for (int d = 0; d < 4; ++d) o[d] = zero8();
  float rowmax[8], rowsum[8];
#pragma unroll
  for (int r = 0; r < 8; ++r) { rowmax[r] = NEG; rowsum[r] = 0.0f; }

  const int ihi = i0 + 15;
  for (int j0 = 0; j0 <= ihi; j0 += 32) {     // causal trim: skip j0 > ihi
    // ---- S = Q K^T over a 16x32 score tile (two 16-col fragments) ----
    v8f s[2];
#pragma unroll
    for (int t = 0; t < 2; ++t) {
      s[t] = zero8();
      const __bf16* krow =
          kbf + (size_t)(b * NSEQ + j0 + t * 16 + ln) * DH + half * 16;
      const v16bf bk0 = *reinterpret_cast<const v16bf*>(krow);       // d 0..31
      const v16bf bk1 = *reinterpret_cast<const v16bf*>(krow + 32);  // d 32..63
      s[t] = wmma_bf16(aq0, bk0, s[t]);
      s[t] = wmma_bf16(aq1, bk1, s[t]);
    }

    // ---- bias + masks + online softmax (row = r + 8*half per C layout) ----
#pragma unroll
    for (int r = 0; r < 8; ++r) {
      const int m = i0 + r + half * 8;
      float sv0, sv1;
      {
        const int n = j0 + ln;
        float v = s[0][r] + bias[((size_t)h * NSEQ + m) * NSEQ + n];
        if (n > m || mask[b * NSEQ + n] == 0) v = NEG;
        sv0 = v;
      }
      {
        const int n = j0 + 16 + ln;
        float v = s[1][r] + bias[((size_t)h * NSEQ + m) * NSEQ + n];
        if (n > m || mask[b * NSEQ + n] == 0) v = NEG;
        sv1 = v;
      }
      float tmax = fmaxf(sv0, sv1);
#pragma unroll
      for (int off = 8; off > 0; off >>= 1)
        tmax = fmaxf(tmax, __shfl_xor(tmax, off, 16));
      const float nm    = fmaxf(rowmax[r], tmax);
      const float rescl = __expf(rowmax[r] - nm);
      rowmax[r] = nm;
      const float p0 = __expf(sv0 - nm);
      const float p1 = __expf(sv1 - nm);
      float ts = p0 + p1;
#pragma unroll
      for (int off = 8; off > 0; off >>= 1) ts += __shfl_xor(ts, off, 16);
      rowsum[r] = rowsum[r] * rescl + ts;
#pragma unroll
      for (int d = 0; d < 4; ++d) o[d][r] = o[d][r] * rescl;
      const int mloc = r + half * 8;
      pls[mloc * 32 + ln]      = (__bf16)p0;
      pls[mloc * 32 + 16 + ln] = (__bf16)p1;
    }
    asm volatile("s_wait_dscnt 0x0" ::: "memory");  // cross-lane LDS visibility

    // ---- O += P V : P re-read from LDS in A layout, V^T gives B frags ----
    const v16bf ap = load_a_frag(&pls[ln * 32], half);
#pragma unroll
    for (int d = 0; d < 4; ++d) {
      const __bf16* vp =
          vt + (size_t)(b * DH + d * 16 + ln) * NSEQ + j0 + half * 16;
      const v16bf bv = *reinterpret_cast<const v16bf*>(vp);
      o[d] = wmma_bf16(ap, bv, o[d]);
    }
  }

  // ---- normalize and store O as bf16 [b][i][h*64+d] ----
#pragma unroll
  for (int r = 0; r < 8; ++r) {
    const float inv = 1.0f / rowsum[r];
    const int m = i0 + r + half * 8;
#pragma unroll
    for (int d = 0; d < 4; ++d) {
      obf[(size_t)(b * NSEQ + m) * INNER + h * DH + d * 16 + ln] =
          (__bf16)(o[d][r] * inv);
    }
  }
}

// ---------------- host orchestration ----------------
extern "C" void kernel_launch(void* const* d_in, const int* in_sizes, int n_in,
                              void* d_out, int out_size, void* d_ws, size_t ws_size,
                              hipStream_t stream) {
  (void)in_sizes; (void)n_in; (void)out_size; (void)ws_size;
  const float*         x     = (const float*)d_in[0];
  const unsigned char* mask  = (const unsigned char*)d_in[1];  // jax bool -> 1B
  const float*         bias  = (const float*)d_in[2];
  const float*         gamma = (const float*)d_in[3];
  const float*         Wq    = (const float*)d_in[4];
  const float*         Wkv   = (const float*)d_in[5];
  const float*         Wo    = (const float*)d_in[6];
  float*               out   = (float*)d_out;

  char* ws = (char*)d_ws;
  auto alloc = [&](size_t bytes) {
    char* p = ws;
    ws += (bytes + 255) & ~(size_t)255;
    return p;
  };
  __bf16* xnbf = (__bf16*)alloc((size_t)ROWS * DIM * 2);
  __bf16* xbf  = (__bf16*)alloc((size_t)ROWS * DIM * 2);
  __bf16* WqT  = (__bf16*)alloc((size_t)INNER * DIM * 2);     // [512][1024]
  __bf16* WkvT = (__bf16*)alloc((size_t)(2 * DH) * DIM * 2);  // [128][1024]
  __bf16* WoT  = (__bf16*)alloc((size_t)DIM * INNER * 2);     // [1024][512]
  __bf16* qbf  = (__bf16*)alloc((size_t)ROWS * INNER * 2);
  __bf16* kbf  = (__bf16*)alloc((size_t)ROWS * DH * 2);
  __bf16* vtbf = (__bf16*)alloc((size_t)BATCH * DH * NSEQ * 2);
  __bf16* obf  = (__bf16*)alloc((size_t)ROWS * INNER * 2);

  // 1) LayerNorm + bf16 casts
  ln_cast_kernel<<<ROWS, 256, 0, stream>>>(x, gamma, xnbf, xbf);

  // 2) weight transposes to bf16 (B operand in [N][K] for contiguous frags)
  transpose_cast_kernel<<<1024, 256, 0, stream>>>(Wq,  WqT,  DIM, INNER);
  transpose_cast_kernel<<<256,  256, 0, stream>>>(Wkv, WkvT, DIM, 2 * DH);
  transpose_cast_kernel<<<1024, 256, 0, stream>>>(Wo,  WoT,  INNER, DIM);

  // 3) projections: Q = LN(x) Wq * SCALE ; [K|V] = x Wkv (V stored transposed)
  gemm_wmma_kernel<0><<<dim3(ROWS / 16, INNER / 64), 32, 0, stream>>>(
      xnbf, WqT, DIM, INNER, qbf, nullptr);
  gemm_wmma_kernel<1><<<dim3(ROWS / 16, (2 * DH) / 64), 32, 0, stream>>>(
      xbf, WkvT, DIM, 2 * DH, kbf, vtbf);

  // 4) flash attention (causal + padding mask + additive bias)
  flash_kernel<<<dim3(NSEQ / 16, HEADS, BATCH), 32, 0, stream>>>(
      qbf, kbf, vtbf, bias, mask, obf);

  // 5) output projection -> f32 d_out
  gemm_wmma_kernel<2><<<dim3(ROWS / 16, DIM / 64), 32, 0, stream>>>(
      obf, WoT, INNER, DIM, out, nullptr);
}
